// NaiveFourierKANLayer_37142877176047
// MI455X (gfx1250) — matrix operand order repaired and checked by reading, not compiled, not run on hardware
//
#include <hip/hip_runtime.h>

#define N_NODES 50000
#define N_EDGES 800000
#define IN_F    32
#define OUT_F   64
#define GRID    8
#define KTOT    512   // 2 * IN_F * GRID  (cos block then sin block)

typedef __attribute__((ext_vector_type(16))) _Float16 v16h;
typedef __attribute__((ext_vector_type(8)))  _Float16 v8h;
typedef __attribute__((ext_vector_type(8)))  float    v8f;

// ---------------------------------------------------------------------------
// Kernel 1: reorder fouriercoeffs [2][64][32][8] -> Wk[kk][j] in f16,
//           kk = d*256 + i*8 + g  (matches basis ordering below).
// ---------------------------------------------------------------------------
__global__ __launch_bounds__(256) void kan_prep_weights(
    const float* __restrict__ fc, _Float16* __restrict__ Wk) {
  int idx = blockIdx.x * 256 + threadIdx.x;        // 0 .. 32767 (exact)
  int j  = idx & 63;
  int kk = idx >> 6;
  int d = kk >> 8, i = (kk >> 3) & 31, g = kk & 7;
  Wk[kk * OUT_F + j] = (_Float16)fc[(((d * OUT_F + j) * IN_F + i) << 3) + g];
}

// ---------------------------------------------------------------------------
// Kernel 2: out[n][j] = bias[j]  (atomic scatter accumulates on top)
// ---------------------------------------------------------------------------
__global__ __launch_bounds__(256) void kan_init_out(
    const float* __restrict__ bias, float* __restrict__ out) {
  int idx = blockIdx.x * 256 + threadIdx.x;
  if (idx < N_NODES * OUT_F) out[idx] = bias[idx & 63];
}

// ---------------------------------------------------------------------------
// Kernel 3: per-node messages via WMMA.
//   Block = 256 threads (8 waves), 32 nodes per block.
//   Phase A: stage x tile (32x32 f32) in LDS.
//   Phase B: build basis tile (32 x 512 f16) in LDS: [cos(g+1)x_i | sin(g+1)x_i]
//   Phase C: each wave computes one 16x16 tile of y = basis * Wk
//            via 16 x v_wmma_f32_16x16x32_f16 (K = 512).
// ---------------------------------------------------------------------------
__global__ __launch_bounds__(256) void kan_node_msg(
    const float* __restrict__ x, const _Float16* __restrict__ Wk,
    float* __restrict__ y, int n_nodes) {
  __shared__ __align__(16) float    xs[32 * IN_F];      // 4 KB
  __shared__ __align__(16) _Float16 basis[32 * KTOT];   // 32 KB

  const int tid   = threadIdx.x;
  const int node0 = blockIdx.x * 32;

  // ---- Phase A: x tile (clamp OOB rows to a valid node; never stored) ----
  for (int idx = tid; idx < 32 * IN_F; idx += 256) {
    int r = idx >> 5, i = idx & 31;
    int n = node0 + r;
    if (n >= n_nodes) n = n_nodes - 1;
    xs[idx] = x[n * IN_F + i];
  }
  __syncthreads();

  // ---- Phase B: Fourier basis; kk = [0,256) cos, [256,512) sin ----
  for (int idx = tid; idx < 32 * 256; idx += 256) {
    int r  = idx >> 8;
    int ig = idx & 255;              // i*8 + g
    int i  = ig >> 3, g = ig & 7;
    float ang = (float)(g + 1) * xs[r * IN_F + i];
    basis[r * KTOT + ig]       = (_Float16)__cosf(ang);
    basis[r * KTOT + 256 + ig] = (_Float16)__sinf(ang);
  }
  __syncthreads();

  // ---- Phase C: WMMA (wave-uniform control flow; EXEC all ones) ----
  const int wave = tid >> 5, lane = tid & 31;
  const int rt = wave >> 2;          // row tile: 0..1  (16 nodes each)
  const int nt = wave & 3;           // col tile: 0..3  (16 out-feats each)
  const int row  = (lane & 15) + rt * 16;      // A-matrix row for this lane
  const int hi   = (lane >> 4) << 3;           // K sub-base: 0 or 8

  v8f acc = {};
  for (int c = 0; c < 16; ++c) {
    // A fragment: elements 0..7 -> K = 32c+hi .. +7 ; 8..15 -> K = +16 more
    const _Float16* pa = &basis[row * KTOT + c * 32 + hi];
    v8h a0 = *(const v8h*)pa;
    v8h a1 = *(const v8h*)(pa + 16);
    v16h a = __builtin_shufflevector(a0, a1, 0, 1, 2, 3, 4, 5, 6, 7,
                                             8, 9, 10, 11, 12, 13, 14, 15);
    // B fragment: N = lane%16; elements 0..7 -> K = 32c+hi .. +7 ; 8..15 -> K = +16 more
    const _Float16* pb = &Wk[(c * 32 + hi) * OUT_F + nt * 16 + (lane & 15)];
    v16h b;
    for (int s = 0; s < 8; ++s) {
      b[s]     = pb[s * OUT_F];
      b[s + 8] = pb[(s + 16) * OUT_F];
    }
    acc = __builtin_amdgcn_wmma_f32_16x16x32_f16(
        /*neg_a=*/false, a, /*neg_b=*/false, b,
        /*c_mod=*/(short)0, acc, /*reuse_a=*/false, /*reuse_b=*/false);
  }

  // ---- Store D: VGPR p -> M = p (lanes 0-15) / p+8 (lanes 16-31) ----
  const int mofs = (lane >> 4) << 3;
  const int col  = (lane & 15) + nt * 16;
  #pragma unroll
  for (int p = 0; p < 8; ++p) {
    int n = node0 + rt * 16 + mofs + p;
    if (n < n_nodes) y[(size_t)n * OUT_F + col] = acc[p];
  }
}

// ---------------------------------------------------------------------------
// Kernel 4: scatter-add messages: out[dst[e]] += y[src[e]].
//   One thread per (edge, 4 feats); float4 gather (L2-resident y),
//   4x global_atomic_add_f32 (no return).
// ---------------------------------------------------------------------------
__global__ __launch_bounds__(256) void kan_scatter(
    const float* __restrict__ y, const int* __restrict__ src,
    const int* __restrict__ dst, float* __restrict__ out) {
  long long idx = (long long)blockIdx.x * 256 + threadIdx.x;
  int e = (int)(idx >> 4);
  if (e >= N_EDGES) return;
  int j = ((int)idx & 15) << 2;
  int s = src[e], d = dst[e];
  const float4 v = *(const float4*)(y + (size_t)s * OUT_F + j);
  float* o = out + (size_t)d * OUT_F + j;
  atomicAdd(o + 0, v.x);
  atomicAdd(o + 1, v.y);
  atomicAdd(o + 2, v.z);
  atomicAdd(o + 3, v.w);
}

// ---------------------------------------------------------------------------
extern "C" void kernel_launch(void* const* d_in, const int* in_sizes, int n_in,
                              void* d_out, int out_size, void* d_ws, size_t ws_size,
                              hipStream_t stream) {
  (void)in_sizes; (void)n_in; (void)out_size; (void)ws_size;
  const float* x    = (const float*)d_in[0];
  const int*   src  = (const int*)d_in[1];
  const int*   dst  = (const int*)d_in[2];
  const float* fc   = (const float*)d_in[3];
  const float* bias = (const float*)d_in[4];
  float* out = (float*)d_out;

  // Workspace layout: y (12,800,000 B, 256-aligned) then Wk (65,536 B f16).
  float*    y  = (float*)d_ws;
  _Float16* Wk = (_Float16*)((char*)d_ws + (size_t)N_NODES * OUT_F * sizeof(float));

  // 1) weight reorder+cvt: 512*64 = 32768 elems = 128 blocks
  kan_prep_weights<<<128, 256, 0, stream>>>(fc, Wk);
  // 2) out = bias
  kan_init_out<<<(N_NODES * OUT_F + 255) / 256, 256, 0, stream>>>(bias, out);
  // 3) node messages (WMMA): 32 nodes/block
  kan_node_msg<<<(N_NODES + 31) / 32, 256, 0, stream>>>(x, Wk, y, N_NODES);
  // 4) edge scatter: E * 16 threads
  kan_scatter<<<(N_EDGES * 16 + 255) / 256, 256, 0, stream>>>(y, src, dst, out);
}
